// RGCNEncoder_71244917506644
// MI455X (gfx1250) — matrix operand order repaired and checked by reading, not compiled, not run on hardware
//
#include <hip/hip_runtime.h>
#include <hip/hip_bf16.h>

namespace {
constexpr int N_NODES  = 50000;
constexpr int N_REL    = 16;
constexpr int EMB      = 128;
constexpr int HID      = 128;
constexpr int N_EDGES  = 1600000;
constexpr int LDS_STRIDE = 136;                 // halves/row: 272B stride, 16B aligned
constexpr int WT_HALVES  = HID * LDS_STRIDE;    // one prepared matrix: 17408 halves (34816 B)
}

typedef __attribute__((ext_vector_type(16))) _Float16 v16h;
typedef __attribute__((ext_vector_type(8)))  _Float16 v8h;
typedef __attribute__((ext_vector_type(8)))  float    v8f;
typedef __attribute__((ext_vector_type(4)))  int      v4i;

// ---------------------------------------------------------------- utilities
__global__ __launch_bounds__(256) void zero_i32_kernel(int* __restrict__ p, int n) {
  int i = blockIdx.x * 256 + threadIdx.x;
  if (i < n) p[i] = 0;
}

__global__ __launch_bounds__(256) void relu_kernel(float* __restrict__ p, int n) {
  int i = blockIdx.x * 256 + threadIdx.x;
  if (i < n) p[i] = fmaxf(p[i], 0.0f);
}

// count edges per (dst, relation) segment — shared by both layers
__global__ __launch_bounds__(256) void count_kernel(const int* __restrict__ dst,
                                                    const int* __restrict__ et,
                                                    int* __restrict__ cnt) {
  int e = blockIdx.x * 256 + threadIdx.x;
  if (e < N_EDGES) atomicAdd(&cnt[dst[e] * N_REL + et[e]], 1);
}

// Pre-convert weights: Wt[mi][o*LDS_STRIDE + i] = (f16) W[mi][i][o]
// (exact padded-transposed image later async-copied straight into LDS)
__global__ __launch_bounds__(256) void wprep_kernel(const float* __restrict__ W,
                                                    _Float16* __restrict__ Wt) {
  int mi = blockIdx.y;
  const float* src = W + (size_t)mi * EMB * HID;
  _Float16*    dst = Wt + (size_t)mi * WT_HALVES;
  for (int idx = blockIdx.x * 256 + threadIdx.x; idx < EMB * HID; idx += gridDim.x * 256) {
    int i = idx >> 7;      // EMB (K) index — coalesced read
    int o = idx & 127;     // HID (N) index
    dst[o * LDS_STRIDE + i] = (_Float16)src[idx];
  }
}

// ---------------------------------------------------------------- WMMA GEMM
// out[n_rows x 128] = x[n_rows x 128] @ W[128 x 128] (+ bias); f16 operands,
// f32 accumulation via v_wmma_f32_16x16x32_f16. Block = 8 waves, 256 rows
// (2 sequential 16-row tiles per wave). Wt is the prepared f16 LDS image.
__global__ __launch_bounds__(256) void gemm_tile_kernel(
    const float*    __restrict__ x,     // [n_rows, EMB]
    const _Float16* __restrict__ Wt,    // [WT_HALVES] prepared image
    const float*    __restrict__ bias,  // [HID] or nullptr
    float*          __restrict__ out,   // [n_rows, HID]
    int n_rows)
{
  __shared__ _Float16 ldsW[WT_HALVES];

#if __has_builtin(__builtin_amdgcn_global_load_async_to_lds_b128)
  {
    constexpr int CHUNKS = (WT_HALVES * 2) / 16;   // 16B chunks = 2176
    for (int idx = threadIdx.x; idx < CHUNKS; idx += 256) {
      __attribute__((address_space(1))) v4i* gp =
          (__attribute__((address_space(1))) v4i*)(Wt + (size_t)idx * 8);
      __attribute__((address_space(3))) v4i* lp =
          (__attribute__((address_space(3))) v4i*)(ldsW + (size_t)idx * 8);
      __builtin_amdgcn_global_load_async_to_lds_b128(gp, lp, 0, 0);
    }
#if __has_builtin(__builtin_amdgcn_s_wait_asynccnt)
    __builtin_amdgcn_s_wait_asynccnt(0);
#else
    asm volatile("s_wait_asynccnt 0" ::: "memory");
#endif
  }
#else
  for (int idx = threadIdx.x; idx < WT_HALVES / 8; idx += 256)
    ((v8h*)ldsW)[idx] = ((const v8h*)Wt)[idx];
#endif
  __syncthreads();

  const int lane = threadIdx.x & 31;
  const int wave = threadIdx.x >> 5;
  const int g    = lane >> 4;   // lane half-group (ISA 16-bit A/B layout)
  const int m    = lane & 15;   // A row within tile == B/D column within tile

#pragma unroll
  for (int t = 0; t < 2; ++t) {
    const int rowBase = blockIdx.x * 256 + t * 128 + wave * 16;
    if (rowBase >= n_rows) continue;   // uniform per wave; EXEC stays full

    v8f acc[8] = {};   // 8 column tiles x (16x16 f32 accumulator)
    const float* xrow = x + (size_t)(rowBase + m) * EMB;

#pragma unroll
    for (int ks = 0; ks < 4; ++ks) {        // K = 128 in steps of 32
      // A fragment (16x32 f16): halves 0..7 -> K = ks*32 + g*8 + j,
      // halves 8..15 -> K = ks*32 + 16 + g*8 + (j-8). Two 32B runs.
      const float4* a4 = (const float4*)(xrow + ks * 32 + g * 8);
      float4 f0 = a4[0], f1 = a4[1];        // run 0
      float4 f2 = a4[4], f3 = a4[5];        // run 1 (+16 floats)
      float af[16] = {f0.x, f0.y, f0.z, f0.w, f1.x, f1.y, f1.z, f1.w,
                      f2.x, f2.y, f2.z, f2.w, f3.x, f3.y, f3.z, f3.w};
      v16h a;
#pragma unroll
      for (int j = 0; j < 16; ++j) a[j] = (_Float16)af[j];

#pragma unroll
      for (int nt = 0; nt < 8; ++nt) {      // 8 column tiles of 16
        // B fragment (32x16 f16): lane holds col nt*16+m, same K order as A
        const _Float16* wrow = &ldsW[(nt * 16 + m) * LDS_STRIDE + ks * 32 + g * 8];
        union { v16h v; v8h h[2]; } bu;
        bu.h[0] = *(const v8h*)(wrow);
        bu.h[1] = *(const v8h*)(wrow + 16);
        acc[nt] = __builtin_amdgcn_wmma_f32_16x16x32_f16(
            false, a, false, bu.v, (short)0, acc[nt], false, false);
      }
    }

    // D layout: VGPR v -> row = rowBase + g*8 + v, col = nt*16 + m
#pragma unroll
    for (int nt = 0; nt < 8; ++nt) {
      int col = nt * 16 + m;
      float bi = bias ? bias[col] : 0.0f;
#pragma unroll
      for (int v = 0; v < 8; ++v) {
        int row = rowBase + g * 8 + v;
        out[(size_t)row * HID + col] = acc[nt][v] + bi;
      }
    }
  }
}

// ---------------------------------------------------------------- edge pass
// For edges of relation `rel`: out[dst] += h_r[src] / max(cnt[dst,rel],1).
// One wave per edge, 4 floats per lane. h_r (25.6MB) is L2-resident.
__global__ __launch_bounds__(256) void edge_agg_kernel(
    const int* __restrict__ src, const int* __restrict__ dst,
    const int* __restrict__ et,  const int* __restrict__ cnt,
    const float* __restrict__ h, float* __restrict__ out, int rel)
{
  int e = blockIdx.x * 8 + (threadIdx.x >> 5);
  int lane = threadIdx.x & 31;
  if (e >= N_EDGES) return;
  if (et[e] != rel) return;
  int s = src[e], d = dst[e];
  float c = (float)cnt[d * N_REL + rel];
  float norm = 1.0f / fmaxf(c, 1.0f);
  const float4 v = *(const float4*)(h + (size_t)s * HID + lane * 4);
  float* o = out + (size_t)d * HID + lane * 4;
  atomicAdd(o + 0, v.x * norm);
  atomicAdd(o + 1, v.y * norm);
  atomicAdd(o + 2, v.z * norm);
  atomicAdd(o + 3, v.w * norm);
}

// ---------------------------------------------------------------- launcher
extern "C" void kernel_launch(void* const* d_in, const int* in_sizes, int n_in,
                              void* d_out, int out_size, void* d_ws, size_t ws_size,
                              hipStream_t stream) {
  (void)in_sizes; (void)n_in; (void)out_size; (void)ws_size;

  const float* node_emb  = (const float*)d_in[0];
  const float* W1        = (const float*)d_in[1];
  const float* root1     = (const float*)d_in[2];
  const float* b1        = (const float*)d_in[3];
  const float* W2        = (const float*)d_in[4];
  const float* root2     = (const float*)d_in[5];
  const float* b2        = (const float*)d_in[6];
  const int*   edge_idx  = (const int*)d_in[7];   // [2, E] int32
  const int*   edge_type = (const int*)d_in[8];   // [E] int32
  const int*   src = edge_idx;
  const int*   dst = edge_idx + N_EDGES;
  float*       out = (float*)d_out;               // [N, HID]

  // workspace layout: counts | h_r | x1 | prepared f16 weights
  char*  ws   = (char*)d_ws;
  int*   cnt  = (int*)ws;                                      // N*R ints (3.2MB)
  float* h    = (float*)(ws + (size_t)N_NODES * N_REL * 4);    // N*HID (25.6MB)
  float* x1   = h + (size_t)N_NODES * HID;                     // N*HID (25.6MB)
  _Float16* wtbase = (_Float16*)(x1 + (size_t)N_NODES * HID);
  _Float16* W1t    = wtbase;                                   // 16 matrices
  _Float16* W2t    = W1t + (size_t)N_REL * WT_HALVES;          // 16 matrices
  _Float16* r1t    = W2t + (size_t)N_REL * WT_HALVES;          // 1 matrix
  _Float16* r2t    = r1t + WT_HALVES;                          // 1 matrix

  const int gemmGrid = (N_NODES + 255) / 256;          // 196
  const int edgeGrid = (N_EDGES + 7) / 8;              // 200000
  const int cntGrid  = (N_NODES * N_REL + 255) / 256;  // 3125
  const int eThGrid  = (N_EDGES + 255) / 256;          // 6250
  const int actGrid  = (N_NODES * HID + 255) / 256;    // 25000

  // prepare padded-transposed f16 weight images (once per call)
  wprep_kernel<<<dim3(16, N_REL), 256, 0, stream>>>(W1, W1t);
  wprep_kernel<<<dim3(16, N_REL), 256, 0, stream>>>(W2, W2t);
  wprep_kernel<<<dim3(16, 1), 256, 0, stream>>>(root1, r1t);
  wprep_kernel<<<dim3(16, 1), 256, 0, stream>>>(root2, r2t);

  // (dst, rel) segment counts — same for both layers
  zero_i32_kernel<<<cntGrid, 256, 0, stream>>>(cnt, N_NODES * N_REL);
  count_kernel<<<eThGrid, 256, 0, stream>>>(dst, edge_type, cnt);

  // ---- layer 1: x1 = relu( agg(node_emb) + node_emb@root1 + b1 )
  gemm_tile_kernel<<<gemmGrid, 256, 0, stream>>>(node_emb, r1t, b1, x1, N_NODES);
  for (int r = 0; r < N_REL; ++r) {
    gemm_tile_kernel<<<gemmGrid, 256, 0, stream>>>(
        node_emb, W1t + (size_t)r * WT_HALVES, nullptr, h, N_NODES);
    edge_agg_kernel<<<edgeGrid, 256, 0, stream>>>(src, dst, edge_type, cnt, h, x1, r);
  }
  relu_kernel<<<actGrid, 256, 0, stream>>>(x1, N_NODES * HID);

  // ---- layer 2: out = agg(x1) + x1@root2 + b2
  gemm_tile_kernel<<<gemmGrid, 256, 0, stream>>>(x1, r2t, b2, out, N_NODES);
  for (int r = 0; r < N_REL; ++r) {
    gemm_tile_kernel<<<gemmGrid, 256, 0, stream>>>(
        x1, W2t + (size_t)r * WT_HALVES, nullptr, h, N_NODES);
    edge_agg_kernel<<<edgeGrid, 256, 0, stream>>>(src, dst, edge_type, cnt, h, out, r);
  }
}
